// VectorQuantizer_42339787604554
// MI455X (gfx1250) — matrix operand order repaired and checked by reading, not compiled, not run on hardware
//
#include <hip/hip_runtime.h>
#include <hip/hip_bf16.h>

// ---------------------------------------------------------------------------
// VQ-VAE vector quantizer for MI455X (gfx1250, wave32, WMMA).
//   x  : (4, 256, 8, 32, 32) f32   -> N=32768 rows of D=256 (d-stride 8192)
//   emb: (1024, 256) f32
// Outputs (concatenated in d_out):
//   [0 .. 8388607]  quantized_st (== emb[argmin]) permuted back to (B,D,T,H,W)
//   [8388608]       vq_loss = 1.25 * mean((q - x)^2)
//   [8388609 ...]   indices as int64 (written as lo/hi u32 pairs, hi = 0)
// Matmul S = x @ emb^T via bf16x3 split (hi*hi + hi*lo + lo*hi), f32 accum,
// v_wmma_f32_16x16x32_bf16. 32 rows/block: each wave computes a 32x16 tile as
// two independent accumulator chains sharing B fragments (2x L2 reuse).
// K-chunk unroll capped at 4 to stay under 256 VGPRs (no MSB mode, keeps
// multi-wave occupancy for hiding scattered B-fragment L2 loads).
// ---------------------------------------------------------------------------

typedef __attribute__((ext_vector_type(16))) __bf16 v16bf;
typedef __attribute__((ext_vector_type(8)))  __bf16 v8bf;
typedef __attribute__((ext_vector_type(8)))  float  v8f;

#define N_TOT   32768
#define DDIM    256
#define KCB     1024
#define M_TILE  32
#define NBLK    (N_TOT / M_TILE)        // 1024
#define AST     264                     // bf16 LDS row stride (pad -> no bank conflicts)
#define FST     260                     // f32  LDS row stride
#define QELEMS  8388608                 // 4*256*8*32*32
#define LOSS_OFF QELEMS

// A fragment (16x32 bf16): lane l holds row M=l%16, half h=l/16;
// element e in [0,16): K = e + 16*h  -> 16 contiguous elts (two 8-elt loads).
__device__ inline v16bf load_a_frag(const __bf16* row, int d0, int hf) {
  union { v16bf v; v8bf h[2]; } u;
  u.h[0] = *(const v8bf*)(row + d0 + 16 * hf);
  u.h[1] = *(const v8bf*)(row + d0 + 16 * hf + 8);
  return u.v;
}

__device__ inline void argmin_tie(float d, int i, float& bd, int& bi) {
  if (d < bd || (d == bd && i < bi)) { bd = d; bi = i; }
}

// ---------------------------------------------------------------------------
// Kernel 1: split codebook into bf16 hi/lo, compute ||e_k||^2.
__global__ __launch_bounds__(256)
void vq_prep(const float* __restrict__ emb, __bf16* __restrict__ embH,
             __bf16* __restrict__ embL, float* __restrict__ enorm) {
  int k = blockIdx.x;
  int d = threadIdx.x;
  size_t off = ((size_t)k << 8) + d;
  float e = emb[off];
  __bf16 h = (__bf16)e;
  __bf16 l = (__bf16)(e - (float)h);
  embH[off] = h;
  embL[off] = l;

  float sq = e * e;
  #pragma unroll
  for (int m = 16; m >= 1; m >>= 1) sq += __shfl_xor(sq, m, 32);
  __shared__ float wsum[8];
  int wave = threadIdx.x >> 5, lane = threadIdx.x & 31;
  if (lane == 0) wsum[wave] = sq;
  __syncthreads();
  if (threadIdx.x == 0) {
    float s = 0.f;
    #pragma unroll
    for (int w = 0; w < 8; ++w) s += wsum[w];
    enorm[k] = s;
  }
}

// ---------------------------------------------------------------------------
// Kernel 2: per 32-row tile: WMMA distance matmul + argmin + gather + store.
__global__ __launch_bounds__(256)
void vq_main(const float* __restrict__ x, const float* __restrict__ emb,
             const __bf16* __restrict__ embH, const __bf16* __restrict__ embL,
             const float* __restrict__ enorm, float* __restrict__ part,
             float* __restrict__ out) {
  __shared__ __bf16 aH[M_TILE * AST];
  __shared__ __bf16 aL[M_TILE * AST];
  __shared__ float  aF[M_TILE * FST];
  __shared__ float  eN[KCB];
  __shared__ float  redD[M_TILE * 8];
  __shared__ int    redI[M_TILE * 8];
  __shared__ int    finI[M_TILE];
  __shared__ float  lsumW[8];

  int t  = threadIdx.x;
  int n0 = blockIdx.x * M_TILE;
  int b  = n0 >> 13;            // batch (8192 spatial per batch; tiles never straddle)
  int s0 = n0 & 8191;
  const float* xb = x + (((size_t)b * DDIM) << 13) + s0;  // xb[d*8192 + r]

  // ---- stage x tile: f32 copy + bf16 hi/lo split; stage ||e||^2 ----
  {
    int r = t & 31, dg = t >> 5;
    for (int d = dg; d < DDIM; d += 8) {
      float v = xb[((size_t)d << 13) + r];
      __bf16 h = (__bf16)v;
      __bf16 l = (__bf16)(v - (float)h);
      aF[r * FST + d] = v;
      aH[r * AST + d] = h;
      aL[r * AST + d] = l;
    }
    for (int k = t; k < KCB; k += 256) eN[k] = enorm[k];
  }
  __syncthreads();

  int wave = t >> 5, lane = t & 31;
  int hf = lane >> 4, l15 = lane & 15;
  const __bf16* aHrow0 = aH + l15 * AST;          // rows 0..15
  const __bf16* aLrow0 = aL + l15 * AST;
  const __bf16* aHrow1 = aH + (l15 + 16) * AST;   // rows 16..31
  const __bf16* aLrow1 = aL + (l15 + 16) * AST;

  float bD0[8], bD1[8];
  int   bI0[8], bI1[8];
  #pragma unroll
  for (int v = 0; v < 8; ++v) {
    bD0[v] = 3.4e38f; bI0[v] = 0;
    bD1[v] = 3.4e38f; bI1[v] = 0;
  }

  // ---- 8 column passes x (8 waves * 16 cols) covers K=1024 ----
  #pragma unroll 1
  for (int iter = 0; iter < 8; ++iter) {
    int c0   = iter * 128 + wave * 16;
    int kcol = c0 + l15;                       // B col owned by this lane
    const __bf16* bHp = embH + ((size_t)kcol << 8);
    const __bf16* bLp = embL + ((size_t)kcol << 8);
    v8f acc0 = {0.f, 0.f, 0.f, 0.f, 0.f, 0.f, 0.f, 0.f};
    v8f acc1 = {0.f, 0.f, 0.f, 0.f, 0.f, 0.f, 0.f, 0.f};
    #pragma unroll 4
    for (int dc = 0; dc < 8; ++dc) {
      int d0 = dc * 32;
      // B element e <-> K = e + 16*half : 16 contiguous bf16 (32B aligned);
      // shared by both row sub-tiles (2x reuse).
      v16bf bh = *(const v16bf*)(bHp + d0 + 16 * hf);
      v16bf bl = *(const v16bf*)(bLp + d0 + 16 * hf);
      v16bf ah0 = load_a_frag(aHrow0, d0, hf);
      v16bf al0 = load_a_frag(aLrow0, d0, hf);
      v16bf ah1 = load_a_frag(aHrow1, d0, hf);
      v16bf al1 = load_a_frag(aLrow1, d0, hf);
      acc0 = __builtin_amdgcn_wmma_f32_16x16x32_bf16(false, ah0, false, bh,
                                                     (short)0, acc0, false, false);
      acc1 = __builtin_amdgcn_wmma_f32_16x16x32_bf16(false, ah1, false, bh,
                                                     (short)0, acc1, false, false);
      acc0 = __builtin_amdgcn_wmma_f32_16x16x32_bf16(false, ah0, false, bl,
                                                     (short)0, acc0, false, false);
      acc1 = __builtin_amdgcn_wmma_f32_16x16x32_bf16(false, ah1, false, bl,
                                                     (short)0, acc1, false, false);
      acc0 = __builtin_amdgcn_wmma_f32_16x16x32_bf16(false, al0, false, bh,
                                                     (short)0, acc0, false, false);
      acc1 = __builtin_amdgcn_wmma_f32_16x16x32_bf16(false, al1, false, bh,
                                                     (short)0, acc1, false, false);
    }
    // dist (up to row-constant ||x||^2) = ||e||^2 - 2*S.
    // kcol strictly increases per lane, so strict '<' keeps lowest index;
    // tie-breaks only needed in the cross-lane/cross-wave reductions.
    float en = eN[kcol];
    #pragma unroll
    for (int v = 0; v < 8; ++v) {
      float d0v = en - 2.0f * acc0[v];
      if (d0v < bD0[v]) { bD0[v] = d0v; bI0[v] = kcol; }
      float d1v = en - 2.0f * acc1[v];
      if (d1v < bD1[v]) { bD1[v] = d1v; bI1[v] = kcol; }
    }
  }

  // ---- argmin reduce across the 16 lanes of each half-wave ----
  #pragma unroll
  for (int m = 8; m >= 1; m >>= 1) {
    #pragma unroll
    for (int v = 0; v < 8; ++v) {
      float od = __shfl_xor(bD0[v], m, 32);
      int   oi = __shfl_xor(bI0[v], m, 32);
      argmin_tie(od, oi, bD0[v], bI0[v]);
      od = __shfl_xor(bD1[v], m, 32);
      oi = __shfl_xor(bI1[v], m, 32);
      argmin_tie(od, oi, bD1[v], bI1[v]);
    }
  }
  if (l15 == 0) {                // lanes 0 (rows m=0..7) and 16 (rows m=8..15)
    #pragma unroll
    for (int v = 0; v < 8; ++v) {
      int row0 = v + 8 * hf;
      redD[row0 * 8 + wave] = bD0[v];
      redI[row0 * 8 + wave] = bI0[v];
      int row1 = 16 + v + 8 * hf;
      redD[row1 * 8 + wave] = bD1[v];
      redI[row1 * 8 + wave] = bI1[v];
    }
  }
  __syncthreads();
  if (t < M_TILE) {              // reduce the 8 wave candidates per row
    float bd = redD[t * 8]; int bi = redI[t * 8];
    #pragma unroll
    for (int w = 1; w < 8; ++w) {
      float od = redD[t * 8 + w]; int oi = redI[t * 8 + w];
      if (od < bd || (od == bd && oi < bi)) { bd = od; bi = oi; }
    }
    finI[t] = bi;
  }
  __syncthreads();

  // ---- gather emb[idx], store quantized (B,D,T,H,W), loss partial ----
  {
    int r = t & 31, dg = t >> 5;
    int idx = finI[r];
    const float* erow = emb + ((size_t)idx << 8);
    float* ob = out + (((size_t)b * DDIM) << 13) + s0 + r;
    float lsum = 0.f;
    for (int d = dg; d < DDIM; d += 8) {
      float q = erow[d];
      ob[(size_t)d << 13] = q;
      float diff = q - aF[r * FST + d];
      lsum += diff * diff;
    }
    #pragma unroll
    for (int m = 16; m >= 1; m >>= 1) lsum += __shfl_xor(lsum, m, 32);
    if (lane == 0) lsumW[wave] = lsum;
    __syncthreads();
    if (t == 0) {
      float s = 0.f;
      #pragma unroll
      for (int w = 0; w < 8; ++w) s += lsumW[w];
      part[blockIdx.x] = s;      // deterministic per-block partial, no atomics
    }
    if (t < M_TILE) {            // indices as int64 (lo=idx, hi=0)
      int n = n0 + t;
      unsigned* idxOut = (unsigned*)(out + LOSS_OFF + 1);
      idxOut[2 * (size_t)n]     = (unsigned)finI[t];
      idxOut[2 * (size_t)n + 1] = 0u;
    }
  }
}

// ---------------------------------------------------------------------------
// Kernel 3: deterministic fixed-order loss reduction.
__global__ void vq_finish(const float* __restrict__ part, float* __restrict__ out) {
  if (threadIdx.x == 0 && blockIdx.x == 0) {
    float s = 0.f;
    for (int i = 0; i < NBLK; ++i) s += part[i];
    // vq_loss = codebook + 0.25*commit = 1.25 * mean((q-x)^2)
    out[LOSS_OFF] = 1.25f * s / (float)QELEMS;
  }
}

// ---------------------------------------------------------------------------
extern "C" void kernel_launch(void* const* d_in, const int* in_sizes, int n_in,
                              void* d_out, int out_size, void* d_ws, size_t ws_size,
                              hipStream_t stream) {
  const float* x   = (const float*)d_in[0];
  const float* emb = (const float*)d_in[1];
  float* out = (float*)d_out;

  char* ws = (char*)d_ws;
  __bf16* embH  = (__bf16*)(ws);                       // 512 KB
  __bf16* embL  = (__bf16*)(ws + 524288);              // 512 KB
  float*  enorm = (float*)(ws + 1048576);              // 4 KB
  float*  part  = (float*)(ws + 1048576 + 4096);       // 4 KB

  vq_prep<<<KCB, 256, 0, stream>>>(emb, embH, embL, enorm);
  vq_main<<<NBLK, 256, 0, stream>>>(x, emb, embH, embL, enorm, part, out);
  vq_finish<<<1, 64, 0, stream>>>(part, out);
}
